// SlotMultiAgentSystem_67018669686856
// MI455X (gfx1250) — matrix operand order repaired and checked by reading, not compiled, not run on hardware
//
#include <hip/hip_runtime.h>
#include <hip/hip_bf16.h>

typedef __attribute__((ext_vector_type(16))) _Float16 v16h;
typedef __attribute__((ext_vector_type(8)))  _Float16 v8h;
typedef __attribute__((ext_vector_type(8)))  float    v8f;

#define LDST 72   // LDS row stride in halves: 144B rows -> 16B aligned, conflict-free over 16 rows

__device__ __forceinline__ v16h make_frag(v8h lo, v8h hi) {
  return __builtin_shufflevector(lo, hi, 0, 1, 2, 3, 4, 5, 6, 7,
                                 8, 9, 10, 11, 12, 13, 14, 15);
}

// CDNA5 async copy: 16 bytes global -> LDS, tracked by ASYNCcnt (no VGPR round-trip).
// lds_addr = low 32 bits of a generic pointer to __shared__ (LDS aperture offset).
__device__ __forceinline__ void async_copy16(unsigned lds_addr, const void* gaddr) {
  asm volatile("global_load_async_to_lds_b128 %0, %1, off"
               :: "v"(lds_addr), "v"(gaddr)
               : "memory");
}
__device__ __forceinline__ void async_wait0() {
  asm volatile("s_wait_asynccnt 0" ::: "memory");
}

// ---------------- weight packing ----------------
// OIHW (O, 64, 3, 3) fp32 -> (9, O, 64) half  [tap][o][c]
__global__ void k_pack_convw(const float* __restrict__ w, _Float16* __restrict__ wp, int O) {
  int gid = blockIdx.x * 256 + threadIdx.x;
  int total = 9 * O * 64;
  if (gid >= total) return;
  int c = gid & 63;
  int o = (gid >> 6) % O;
  int t = gid / (O * 64);
  int ky = t / 3, kx = t % 3;
  wp[gid] = (_Float16)w[((o * 64 + c) * 3 + ky) * 3 + kx];
}

// wk(256,64), wv(256,64) fp32 -> (128, 256) half, [n][k] pre-transposed
__global__ void k_pack_kv(const float* __restrict__ wk, const float* __restrict__ wv,
                          _Float16* __restrict__ kvp) {
  int gid = blockIdx.x * 256 + threadIdx.x;  // 32768
  int n = gid >> 8, k = gid & 255;
  float v = (n < 64) ? wk[k * 64 + n] : wv[k * 64 + (n - 64)];
  kvp[gid] = (_Float16)v;
}

// ---------------- conv0 (3->64, scalar, K=27) ----------------
__global__ void k_conv0(const float* __restrict__ img, const float* __restrict__ w,
                        const float* __restrict__ b, _Float16* __restrict__ out) {
  int gid = blockIdx.x * 256 + threadIdx.x;  // 32*32*32*64
  int o = gid & 63;
  int p = gid >> 6;
  int x = p & 31, y = (p >> 5) & 31, bb = p >> 10;
  float s = b[o];
  for (int c = 0; c < 3; ++c)
    for (int ky = 0; ky < 3; ++ky) {
      int sy = y + ky - 1; if (sy < 0 || sy > 31) continue;
      for (int kx = 0; kx < 3; ++kx) {
        int sx = x + kx - 1; if (sx < 0 || sx > 31) continue;
        s += img[((bb * 3 + c) * 32 + sy) * 32 + sx] * w[((o * 3 + c) * 3 + ky) * 3 + kx];
      }
    }
  out[(size_t)p * 64 + o] = (_Float16)fmaxf(s, 0.f);
}

// ---------------- implicit-GEMM 3x3 conv via WMMA ----------------
// act: NHWC half (32,32,32,64); wp: (9, cout, 64) half; out: NHWC half (., cout)
__global__ void k_conv_wmma(const _Float16* __restrict__ act, const _Float16* __restrict__ wp,
                            const float* __restrict__ bias, _Float16* __restrict__ out,
                            int cout, int relu) {
  __shared__ alignas(16) _Float16 As[128 * LDST];
  __shared__ alignas(16) _Float16 Bs[64 * LDST];
  int tid = threadIdx.x;
  int wid = tid >> 5, lane = tid & 31;
  int wm = wid >> 1, wn = wid & 1;       // 4x2 wave grid
  int lh = lane >> 4, lm = lane & 15;
  int m0 = blockIdx.x * 128;
  int n0 = blockIdx.y * 64;
  v8f acc[2][2] = {};

  // per-thread staging coordinates (invariant over tap loop)
  int ar = tid >> 1;                 // A row 0..127
  int ac0 = (tid & 1) * 32;          // A col 0 or 32
  int ap = m0 + ar;
  int abb = ap >> 10, ayy = (ap >> 5) & 31, axx = ap & 31;
  unsigned a_lds = (unsigned)(uintptr_t)&As[ar * LDST + ac0];
  int bn = tid >> 2;                 // B row 0..63
  int bc0 = (tid & 3) * 16;
  unsigned b_lds = (unsigned)(uintptr_t)&Bs[bn * LDST + bc0];

  for (int t = 0; t < 9; ++t) {
    int ky = t / 3 - 1, kx = t % 3 - 1;
    __syncthreads();
    {  // A tile: 128 spatial rows x 64 ch, zero-padded borders
      int sy = ayy + ky, sx = axx + kx;
      if ((unsigned)sy < 32u && (unsigned)sx < 32u) {
        const _Float16* src = act + (((size_t)(abb * 32 + sy) * 32 + sx) * 64 + ac0);
#pragma unroll
        for (int i = 0; i < 4; ++i) async_copy16(a_lds + i * 16, src + i * 8);
      } else {
        v8h z = {};
        v8h* d8 = (v8h*)&As[ar * LDST + ac0];
        d8[0] = z; d8[1] = z; d8[2] = z; d8[3] = z;
      }
    }
    {  // B tile: 64 out-ch rows x 64 in-ch (pre-transposed pack)
      const _Float16* src = wp + ((size_t)(t * cout + n0 + bn)) * 64 + bc0;
      async_copy16(b_lds, src);
      async_copy16(b_lds + 16, src + 8);
    }
    async_wait0();
    __syncthreads();
#pragma unroll
    for (int kk = 0; kk < 64; kk += 32) {
      v16h afr[2], bfr[2];
#pragma unroll
      for (int i = 0; i < 2; ++i) {  // A frag: lane<16 -> K {0..7,16..23}; lane>=16 -> +8
        const _Float16* base = &As[(wm * 32 + i * 16 + lm) * LDST + kk + lh * 8];
        afr[i] = make_frag(*(const v8h*)base, *(const v8h*)(base + 16));
      }
#pragma unroll
      for (int j = 0; j < 2; ++j) {  // B frag: lane<16 -> K {0..7,16..23}; lane>=16 -> +8, n = lm
        const _Float16* base = &Bs[(wn * 32 + j * 16 + lm) * LDST + kk + lh * 8];
        bfr[j] = make_frag(*(const v8h*)base, *(const v8h*)(base + 16));
      }
#pragma unroll
      for (int i = 0; i < 2; ++i)
#pragma unroll
        for (int j = 0; j < 2; ++j)
          acc[i][j] = __builtin_amdgcn_wmma_f32_16x16x32_f16(
              false, afr[i], false, bfr[j], (short)0, acc[i][j], false, false);
    }
  }
#pragma unroll
  for (int i = 0; i < 2; ++i)
#pragma unroll
    for (int j = 0; j < 2; ++j) {
      int n = n0 + wn * 32 + j * 16 + lm;
      float bv = bias[n];
#pragma unroll
      for (int r = 0; r < 8; ++r) {
        int m = m0 + wm * 32 + i * 16 + lh * 8 + r;  // C/D layout: m = r + 8*(lane>=16)
        float val = acc[i][j][r] + bv;
        if (relu) val = fmaxf(val, 0.f);
        out[(size_t)m * cout + n] = (_Float16)val;
      }
    }
}

// ---------------- LayerNorm over 256 channels (one wave per row) ----------------
__global__ void k_ln(const _Float16* __restrict__ feats, _Float16* __restrict__ xn) {
  int row = blockIdx.x * 8 + (threadIdx.x >> 5);
  int lane = threadIdx.x & 31;
  const _Float16* src = feats + (size_t)row * 256 + lane * 8;
  float v[8];
#pragma unroll
  for (int i = 0; i < 8; ++i) v[i] = (float)src[i];
  float s = 0.f;
#pragma unroll
  for (int i = 0; i < 8; ++i) s += v[i];
#pragma unroll
  for (int m = 16; m >= 1; m >>= 1) s += __shfl_xor(s, m, 32);
  float mean = s * (1.f / 256.f);
  float q = 0.f;
#pragma unroll
  for (int i = 0; i < 8; ++i) { float c = v[i] - mean; q += c * c; }
#pragma unroll
  for (int m = 16; m >= 1; m >>= 1) q += __shfl_xor(q, m, 32);
  float inv = rsqrtf(q * (1.f / 256.f) + 1e-5f);
  _Float16* dst = xn + (size_t)row * 256 + lane * 8;
#pragma unroll
  for (int i = 0; i < 8; ++i) dst[i] = (_Float16)((v[i] - mean) * inv);
}

// ---------------- k|v projection GEMM via WMMA ----------------
// A: (32768, 256) half; Bp: (128, 256) half [n][k]; out: k(.,64) v(.,64) fp32
__global__ void k_gemm_kv(const _Float16* __restrict__ A, const _Float16* __restrict__ Bp,
                          float* __restrict__ kout, float* __restrict__ vout) {
  __shared__ alignas(16) _Float16 As[128 * LDST];
  __shared__ alignas(16) _Float16 Bs[64 * LDST];
  int tid = threadIdx.x;
  int wid = tid >> 5, lane = tid & 31;
  int wm = wid >> 1, wn = wid & 1;
  int lh = lane >> 4, lm = lane & 15;
  int m0 = blockIdx.x * 128;
  int n0 = blockIdx.y * 64;
  v8f acc[2][2] = {};

  int ar = tid >> 1;  int ac0 = (tid & 1) * 32;
  int bn = tid >> 2;  int bc0 = (tid & 3) * 16;
  unsigned a_lds = (unsigned)(uintptr_t)&As[ar * LDST + ac0];
  unsigned b_lds = (unsigned)(uintptr_t)&Bs[bn * LDST + bc0];

  for (int kb = 0; kb < 256; kb += 64) {
    __syncthreads();
    {
      const _Float16* src = A + (size_t)(m0 + ar) * 256 + kb + ac0;
#pragma unroll
      for (int i = 0; i < 4; ++i) async_copy16(a_lds + i * 16, src + i * 8);
    }
    {
      const _Float16* src = Bp + (size_t)(n0 + bn) * 256 + kb + bc0;
      async_copy16(b_lds, src);
      async_copy16(b_lds + 16, src + 8);
    }
    async_wait0();
    __syncthreads();
#pragma unroll
    for (int kk = 0; kk < 64; kk += 32) {
      v16h afr[2], bfr[2];
#pragma unroll
      for (int i = 0; i < 2; ++i) {
        const _Float16* base = &As[(wm * 32 + i * 16 + lm) * LDST + kk + lh * 8];
        afr[i] = make_frag(*(const v8h*)base, *(const v8h*)(base + 16));
      }
#pragma unroll
      for (int j = 0; j < 2; ++j) {
        const _Float16* base = &Bs[(wn * 32 + j * 16 + lm) * LDST + kk + lh * 8];
        bfr[j] = make_frag(*(const v8h*)base, *(const v8h*)(base + 16));
      }
#pragma unroll
      for (int i = 0; i < 2; ++i)
#pragma unroll
        for (int j = 0; j < 2; ++j)
          acc[i][j] = __builtin_amdgcn_wmma_f32_16x16x32_f16(
              false, afr[i], false, bfr[j], (short)0, acc[i][j], false, false);
    }
  }
#pragma unroll
  for (int i = 0; i < 2; ++i)
#pragma unroll
    for (int j = 0; j < 2; ++j) {
      int n = n0 + wn * 32 + j * 16 + lm;
#pragma unroll
      for (int r = 0; r < 8; ++r) {
        int m = m0 + wm * 32 + i * 16 + lh * 8 + r;
        float val = acc[i][j][r];
        if (n < 64) kout[(size_t)m * 64 + n] = val;
        else        vout[(size_t)m * 64 + (n - 64)] = val;
      }
    }
}

// ---------------- slot attention pieces ----------------
__global__ void k_slots_init(const float* __restrict__ mu, const float* __restrict__ ls,
                             const float* __restrict__ noise, float* __restrict__ slots) {
  int gid = blockIdx.x * 256 + threadIdx.x;
  if (gid >= 14336) return;
  int sd = gid % 448;
  slots[gid] = mu[sd] + noise[gid] * expf(ls[sd]);
}

__global__ void k_q(const float* __restrict__ slots, const float* __restrict__ wq,
                    float* __restrict__ q, float* __restrict__ sums) {
  int bs = blockIdx.x;          // 0..223
  int d = threadIdx.x;          // 0..63
  __shared__ float red[64];
  float v = slots[bs * 64 + d];
  red[d] = v; __syncthreads();
  for (int off = 32; off > 0; off >>= 1) { if (d < off) red[d] += red[d + off]; __syncthreads(); }
  float mean = red[0] * (1.f / 64.f); __syncthreads();
  float c = v - mean; red[d] = c * c; __syncthreads();
  for (int off = 32; off > 0; off >>= 1) { if (d < off) red[d] += red[d + off]; __syncthreads(); }
  float inv = rsqrtf(red[0] * (1.f / 64.f) + 1e-5f); __syncthreads();
  red[d] = (v - mean) * inv;   // normalized slot
  __syncthreads();
  float acc = 0.f;
  for (int e = 0; e < 64; ++e) acc += red[e] * wq[e * 64 + d];
  q[bs * 64 + d] = acc * 0.125f;        // * 64^-0.5
  if (d == 0) sums[bs] = 0.f;           // reset per-iteration attn column sums
}

__global__ void k_attn(const float* __restrict__ kbuf, const float* __restrict__ q,
                       float* __restrict__ attn, float* __restrict__ sums) {
  __shared__ float qs[448];
  __shared__ float sred[7];
  int p = blockIdx.x * 256 + threadIdx.x;  // 0..32767, b uniform per block
  int b = p >> 10;
  if (threadIdx.x < 7) sred[threadIdx.x] = 0.f;
  for (int i = threadIdx.x; i < 448; i += 256) qs[i] = q[b * 448 + i];
  __syncthreads();
  const float* kr = kbuf + (size_t)p * 64;
  float lg[7] = {0.f, 0.f, 0.f, 0.f, 0.f, 0.f, 0.f};
  for (int d = 0; d < 64; ++d) {
    float kd = kr[d];
#pragma unroll
    for (int s = 0; s < 7; ++s) lg[s] += kd * qs[s * 64 + d];
  }
  float mx = lg[0];
#pragma unroll
  for (int s = 1; s < 7; ++s) mx = fmaxf(mx, lg[s]);
  float se = 0.f;
#pragma unroll
  for (int s = 0; s < 7; ++s) { lg[s] = expf(lg[s] - mx); se += lg[s]; }
  float inv = 1.f / se;
#pragma unroll
  for (int s = 0; s < 7; ++s) {
    float a = lg[s] * inv + 1e-8f;
    attn[(size_t)p * 7 + s] = a;
    atomicAdd(&sred[s], a);
  }
  __syncthreads();
  if (threadIdx.x < 7) atomicAdd(&sums[b * 7 + threadIdx.x], sred[threadIdx.x]);
}

__global__ void k_updates(const float* __restrict__ attn, const float* __restrict__ sums,
                          const float* __restrict__ vbuf, float* __restrict__ updates) {
  int bs = blockIdx.x;            // b*7+s
  int b = bs / 7, s = bs % 7;
  int d = threadIdx.x;
  float acc = 0.f;
  const float* ab = attn + (size_t)b * 1024 * 7 + s;
  const float* vb = vbuf + (size_t)b * 1024 * 64 + d;
  for (int n = 0; n < 1024; ++n) acc += ab[n * 7] * vb[n * 64];
  updates[bs * 64 + d] = acc / sums[bs];
}

__global__ void k_gru_mlp(float* __restrict__ slots, const float* __restrict__ updates,
                          const float* __restrict__ gru_wi, const float* __restrict__ gru_wh,
                          const float* __restrict__ gru_b,
                          const float* __restrict__ mlp_w1, const float* __restrict__ mlp_b1,
                          const float* __restrict__ mlp_w2, const float* __restrict__ mlp_b2) {
  int bs = blockIdx.x;
  int d = threadIdx.x;
  __shared__ float prev[64], upd[64], sl[64], hid[128], red[64];
  prev[d] = slots[bs * 64 + d];
  upd[d] = updates[bs * 64 + d];
  __syncthreads();
  float gi_z = gru_b[d], gi_r = gru_b[64 + d], gi_n = gru_b[128 + d];
  float gh_z = 0.f, gh_r = 0.f, gh_n = 0.f;
  for (int e = 0; e < 64; ++e) {
    float u = upd[e], p = prev[e];
    gi_z += u * gru_wi[e * 192 + d];
    gi_r += u * gru_wi[e * 192 + 64 + d];
    gi_n += u * gru_wi[e * 192 + 128 + d];
    gh_z += p * gru_wh[e * 192 + d];
    gh_r += p * gru_wh[e * 192 + 64 + d];
    gh_n += p * gru_wh[e * 192 + 128 + d];
  }
  float z = 1.f / (1.f + expf(-(gi_z + gh_z)));
  float r = 1.f / (1.f + expf(-(gi_r + gh_r)));
  float n = tanhf(gi_n + r * gh_n);
  float s = (1.f - z) * n + z * prev[d];
  sl[d] = s; red[d] = s;
  __syncthreads();
  for (int off = 32; off > 0; off >>= 1) { if (d < off) red[d] += red[d + off]; __syncthreads(); }
  float mean = red[0] * (1.f / 64.f); __syncthreads();
  float c = s - mean; red[d] = c * c; __syncthreads();
  for (int off = 32; off > 0; off >>= 1) { if (d < off) red[d] += red[d + off]; __syncthreads(); }
  float inv = rsqrtf(red[0] * (1.f / 64.f) + 1e-5f); __syncthreads();
  red[d] = (s - mean) * inv;
  __syncthreads();
  for (int jj = 0; jj < 2; ++jj) {
    int j = d * 2 + jj;
    float h = mlp_b1[j];
    for (int e = 0; e < 64; ++e) h += red[e] * mlp_w1[e * 128 + j];
    hid[j] = fmaxf(h, 0.f);
  }
  __syncthreads();
  float o = mlp_b2[d];
  for (int j = 0; j < 128; ++j) o += hid[j] * mlp_w2[j * 64 + d];
  slots[bs * 64 + d] = s + o;
}

// ---------------- VAE-estimator scores ----------------
__global__ void k_scores(const float* __restrict__ slots, const float* __restrict__ est_we,
                         const float* __restrict__ est_wd, float* __restrict__ scores) {
  __shared__ float sl[64], lat[32], red[64];
  int a = blockIdx.x % 50;
  int bs = blockIdx.x / 50;
  int d = threadIdx.x;
  sl[d] = slots[bs * 64 + d];
  __syncthreads();
  if (d < 32) {
    float l = 0.f;
    for (int e = 0; e < 64; ++e) l += sl[e] * est_we[((size_t)a * 64 + e) * 32 + d];
    lat[d] = l;
  }
  __syncthreads();
  float rec = 0.f;
  for (int z = 0; z < 32; ++z) rec += lat[z] * est_wd[((size_t)a * 32 + z) * 64 + d];
  float diff = rec - sl[d];
  red[d] = diff * diff;
  __syncthreads();
  for (int off = 32; off > 0; off >>= 1) { if (d < off) red[d] += red[d + off]; __syncthreads(); }
  if (d == 0) scores[blockIdx.x] = -red[0];
}

__global__ void k_topk(const float* __restrict__ scores, int* __restrict__ idx) {
  int t = blockIdx.x * 256 + threadIdx.x;
  if (t >= 224) return;
  const float* sc = scores + t * 50;
  int i1 = -1, i2 = -1, i3 = -1;
  float b1 = -3.4e38f, b2 = -3.4e38f, b3 = -3.4e38f;
  for (int i = 0; i < 50; ++i) { float v = sc[i]; if (v > b1) { b1 = v; i1 = i; } }
  for (int i = 0; i < 50; ++i) { if (i == i1) continue; float v = sc[i]; if (v > b2) { b2 = v; i2 = i; } }
  for (int i = 0; i < 50; ++i) { if (i == i1 || i == i2) continue; float v = sc[i]; if (v > b3) { b3 = v; i3 = i; } }
  idx[t * 3 + 0] = i1; idx[t * 3 + 1] = i2; idx[t * 3 + 2] = i3;
}

// ---------------- expert (residual MLP) dispatch ----------------
__global__ void k_expert(const float* __restrict__ slots, const int* __restrict__ idx,
                         const float* __restrict__ win, const float* __restrict__ bin,
                         const float* __restrict__ wblk, const float* __restrict__ bblk,
                         const float* __restrict__ wout, const float* __restrict__ bout,
                         float* __restrict__ out) {
  __shared__ float tok[64];
  __shared__ float h[256];
  int t = blockIdx.x;                    // (b*7+s)*3+kk
  int b = t / 21, rem = t % 21, s = rem / 3, kk = rem % 3;
  int a = idx[t];
  int g = threadIdx.x;
  if (g < 64) tok[g] = slots[(b * 7 + s) * 64 + g];
  __syncthreads();
  float acc = bin[a * 256 + g];
  for (int e = 0; e < 64; ++e) acc += tok[e] * win[((size_t)a * 64 + e) * 256 + g];
  h[g] = fmaxf(acc, 0.f);
  __syncthreads();
  for (int i = 0; i < 3; ++i) {
    const float* wb = wblk + ((size_t)(a * 3 + i)) * 65536;
    __builtin_prefetch(wb + g, 0, 3);
    float acc2 = h[g] + bblk[(a * 3 + i) * 256 + g];
    for (int e = 0; e < 256; ++e) acc2 += h[e] * wb[e * 256 + g];
    acc2 = fmaxf(acc2, 0.f);
    __syncthreads();
    h[g] = acc2;
    __syncthreads();
  }
  if (g < 128) {
    float o = bout[a * 128 + g];
    for (int e = 0; e < 256; ++e) o += h[e] * wout[((size_t)a * 256 + e) * 128 + g];
    out[b * 2688 + (s * 3 + kk) * 128 + g] = o;
  }
}

extern "C" void kernel_launch(void* const* d_in, const int* in_sizes, int n_in,
                              void* d_out, int out_size, void* d_ws, size_t ws_size,
                              hipStream_t stream) {
  (void)in_sizes; (void)n_in; (void)out_size; (void)ws_size;
  const float* images   = (const float*)d_in[0];
  const float* noise    = (const float*)d_in[1];
  const float* enc_w0   = (const float*)d_in[2];
  const float* enc_b0   = (const float*)d_in[3];
  const float* enc_w    = (const float*)d_in[4];
  const float* enc_b    = (const float*)d_in[5];
  const float* enc_wo   = (const float*)d_in[6];
  const float* enc_bo   = (const float*)d_in[7];
  const float* wk       = (const float*)d_in[8];
  const float* wv       = (const float*)d_in[9];
  const float* wq       = (const float*)d_in[10];
  const float* gru_wi   = (const float*)d_in[11];
  const float* gru_wh   = (const float*)d_in[12];
  const float* gru_b    = (const float*)d_in[13];
  const float* mlp_w1   = (const float*)d_in[14];
  const float* mlp_b1   = (const float*)d_in[15];
  const float* mlp_w2   = (const float*)d_in[16];
  const float* mlp_b2   = (const float*)d_in[17];
  const float* slots_mu = (const float*)d_in[18];
  const float* slots_ls = (const float*)d_in[19];
  const float* est_we   = (const float*)d_in[20];
  const float* est_wd   = (const float*)d_in[21];
  const float* ag_win   = (const float*)d_in[22];
  const float* ag_bin   = (const float*)d_in[23];
  const float* ag_wblk  = (const float*)d_in[24];
  const float* ag_bblk  = (const float*)d_in[25];
  const float* ag_wout  = (const float*)d_in[26];
  const float* ag_bout  = (const float*)d_in[27];
  float* outp = (float*)d_out;

  char* base = (char*)d_ws;
  size_t off = 0;
  auto take = [&](size_t bytes) -> char* {
    char* p = base + off;
    off += (bytes + 255) & ~(size_t)255;
    return p;
  };
  _Float16* act0  = (_Float16*)take(2097152 * 2);   // conv0 out NHWC half
  _Float16* actA  = (_Float16*)take(2097152 * 2);
  _Float16* actB  = (_Float16*)take(2097152 * 2);
  _Float16* feats = (_Float16*)take((size_t)8388608 * 2);
  _Float16* xn    = (_Float16*)take((size_t)8388608 * 2);
  float*    kbuf  = (float*)take((size_t)2097152 * 4);
  float*    vbuf  = (float*)take((size_t)2097152 * 4);
  _Float16* w1p   = (_Float16*)take(36864 * 2);
  _Float16* w2p   = (_Float16*)take(36864 * 2);
  _Float16* w3p   = (_Float16*)take(36864 * 2);
  _Float16* wop   = (_Float16*)take(147456 * 2);
  _Float16* kvp   = (_Float16*)take(32768 * 2);
  float* slots    = (float*)take(14336 * 4);
  float* qbuf     = (float*)take(14336 * 4);
  float* attn     = (float*)take(229376 * 4);
  float* sums     = (float*)take(224 * 4);
  float* updates  = (float*)take(14336 * 4);
  float* scores   = (float*)take(11200 * 4);
  int*   idxbuf   = (int*)take(672 * 4);

  // weight packing (fp32 -> half, fragment-friendly layouts)
  k_pack_convw<<<144, 256, 0, stream>>>(enc_w + 0 * 36864, w1p, 64);
  k_pack_convw<<<144, 256, 0, stream>>>(enc_w + 1 * 36864, w2p, 64);
  k_pack_convw<<<144, 256, 0, stream>>>(enc_w + 2 * 36864, w3p, 64);
  k_pack_convw<<<576, 256, 0, stream>>>(enc_wo, wop, 256);
  k_pack_kv<<<128, 256, 0, stream>>>(wk, wv, kvp);

  // encoder
  k_conv0<<<8192, 256, 0, stream>>>(images, enc_w0, enc_b0, act0);
  k_conv_wmma<<<dim3(256, 1), 256, 0, stream>>>(act0, w1p, enc_b + 0, actA, 64, 1);
  k_conv_wmma<<<dim3(256, 1), 256, 0, stream>>>(actA, w2p, enc_b + 64, actB, 64, 1);
  k_conv_wmma<<<dim3(256, 1), 256, 0, stream>>>(actB, w3p, enc_b + 128, actA, 64, 1);
  k_conv_wmma<<<dim3(256, 4), 256, 0, stream>>>(actA, wop, enc_bo, feats, 256, 0);
  k_ln<<<4096, 256, 0, stream>>>(feats, xn);
  k_gemm_kv<<<dim3(256, 2), 256, 0, stream>>>(xn, kvp, kbuf, vbuf);

  // slot attention
  k_slots_init<<<56, 256, 0, stream>>>(slots_mu, slots_ls, noise, slots);
  for (int it = 0; it < 3; ++it) {
    k_q<<<224, 64, 0, stream>>>(slots, wq, qbuf, sums);
    k_attn<<<128, 256, 0, stream>>>(kbuf, qbuf, attn, sums);
    k_updates<<<224, 64, 0, stream>>>(attn, sums, vbuf, updates);
    k_gru_mlp<<<224, 64, 0, stream>>>(slots, updates, gru_wi, gru_wh, gru_b,
                                      mlp_w1, mlp_b1, mlp_w2, mlp_b2);
  }

  // scoring + routing + experts
  k_scores<<<11200, 64, 0, stream>>>(slots, est_we, est_wd, scores);
  k_topk<<<1, 256, 0, stream>>>(scores, idxbuf);
  k_expert<<<672, 256, 0, stream>>>(slots, idxbuf, ag_win, ag_bin, ag_wblk, ag_bblk,
                                    ag_wout, ag_bout, outp);
}
